// my_model_27006754357265
// MI455X (gfx1250) — matrix operand order repaired and checked by reading, not compiled, not run on hardware
//
#include <hip/hip_runtime.h>

typedef __attribute__((ext_vector_type(16))) _Float16 v16h;
typedef __attribute__((ext_vector_type(8)))  float    v8f;

#define WMMA_F16(a, b, c) \
  __builtin_amdgcn_wmma_f32_16x16x32_f16(false, (a), false, (b), (short)0, (c), false, false)

// Model dims (fixed by reference): B=4, T=512, I=128, H=128, 4H=512, BT=2048
#define BT_ROWS 2048
#define TSTEPS  512
#define HDIM    128
#define G4H     512

// ---------------- fast activations (v_exp_f32 based) ----------------
__device__ __forceinline__ float tanh_fast(float x) {
  x = fminf(fmaxf(x, -15.f), 15.f);
  float e = __expf(2.f * x);
  return (e - 1.f) / (e + 1.f);
}
__device__ __forceinline__ float sigm(float x) {
  return 1.f / (1.f + __expf(-x));
}

// ---------------- WMMA tile loaders (wave32 layouts, f32 -> f16 in flight) ---
// A 16x32 f16: lane L: M = L%16, half = L/16; a[j]: K = j + 16*half
__device__ __forceinline__ v16h load_a_f32(const float* A, int lda, int m0, int k0) {
  int lane = threadIdx.x & 31;
  int m = lane & 15, half = lane >> 4;
  const float* p = A + (size_t)(m0 + m) * lda + k0 + 16 * half;
  v16h a;
#pragma unroll
  for (int j = 0; j < 16; ++j) a[j] = (_Float16)p[j];
  return a;
}
// B 32x16 f16 from W row-major [N,K] (i.e. C = A @ W^T): lane: N=L%16, K=j+16*half
__device__ __forceinline__ v16h load_bT_f32(const float* W, int ldw, int n0, int k0) {
  int lane = threadIdx.x & 31;
  int n = lane & 15, half = lane >> 4;
  const float* p = W + (size_t)(n0 + n) * ldw + k0 + 16 * half;
  v16h b;
#pragma unroll
  for (int j = 0; j < 16; ++j) b[j] = (_Float16)p[j];
  return b;
}
// B 32x16 f16 from W row-major [K,N] (i.e. C = A @ W)
__device__ __forceinline__ v16h load_bN_f32(const float* W, int ldw, int n0, int k0) {
  int lane = threadIdx.x & 31;
  int n = lane & 15, half = lane >> 4;
  const float* p = W + (size_t)(k0 + 16 * half) * ldw + n0 + n;
  v16h b;
#pragma unroll
  for (int j = 0; j < 16; ++j) b[j] = (_Float16)p[(size_t)j * ldw];
  return b;
}
// A 16x32 f16 from LDS h buffer [16][128]
__device__ __forceinline__ v16h load_a_lds(const _Float16 (*hbuf)[HDIM], int k0) {
  int lane = threadIdx.x & 31;
  int m = lane & 15, half = lane >> 4;
  const _Float16* p = &hbuf[m][k0 + 16 * half];
  v16h a;
#pragma unroll
  for (int j = 0; j < 16; ++j) a[j] = p[j];
  return a;
}

// ---------------- generic WMMA GEMM: C = A1@B1 (+ A2@B2) + bias --------------
// BT=true: B given as W row-major [N,K] (C=A@W^T); BT=false: W row-major [K,N].
// Each wave computes a 16x32 output slab (two 16x16 tiles sharing one A tile);
// 8 waves per block. M mult of 16, N mult of 32, K mult of 32.
template <bool BT>
__global__ __launch_bounds__(256) void gemm_wmma(
    const float* __restrict__ A1, const float* __restrict__ B1, int K1,
    const float* __restrict__ A2, const float* __restrict__ B2, int K2,
    const float* __restrict__ bias, float* __restrict__ C,
    int M, int N, int lda1, int ldb1, int lda2, int ldb2, int ldc) {
  int wid = blockIdx.x * 8 + (threadIdx.x >> 5);
  int ntiles = N >> 5;  // 32-wide slabs
  int mt = wid / ntiles, nt = wid % ntiles;
  if (mt >= (M >> 4)) return;  // uniform per wave
  int m0 = mt << 4, n0 = nt << 5;
  int lane = threadIdx.x & 31;
  int nn = lane & 15, half = lane >> 4;

  v8f acc0, acc1;
  float bv0 = bias ? bias[n0 + nn] : 0.f;
  float bv1 = bias ? bias[n0 + 16 + nn] : 0.f;
#pragma unroll
  for (int r = 0; r < 8; ++r) { acc0[r] = bv0; acc1[r] = bv1; }

  for (int k0 = 0; k0 < K1; k0 += 32) {
    v16h a  = load_a_f32(A1, lda1, m0, k0);
    v16h b0 = BT ? load_bT_f32(B1, ldb1, n0, k0)      : load_bN_f32(B1, ldb1, n0, k0);
    v16h b1 = BT ? load_bT_f32(B1, ldb1, n0 + 16, k0) : load_bN_f32(B1, ldb1, n0 + 16, k0);
    acc0 = WMMA_F16(a, b0, acc0);
    acc1 = WMMA_F16(a, b1, acc1);
  }
  if (A2) {
    for (int k0 = 0; k0 < K2; k0 += 32) {
      v16h a  = load_a_f32(A2, lda2, m0, k0);
      v16h b0 = BT ? load_bT_f32(B2, ldb2, n0, k0)      : load_bN_f32(B2, ldb2, n0, k0);
      v16h b1 = BT ? load_bT_f32(B2, ldb2, n0 + 16, k0) : load_bN_f32(B2, ldb2, n0 + 16, k0);
      acc0 = WMMA_F16(a, b0, acc0);
      acc1 = WMMA_F16(a, b1, acc1);
    }
  }
#pragma unroll
  for (int r = 0; r < 8; ++r) {
    C[(size_t)(m0 + r + 8 * half) * ldc + n0 + nn]      = acc0[r];
    C[(size_t)(m0 + r + 8 * half) * ldc + n0 + 16 + nn] = acc1[r];
  }
}

// ---------------- LSTM scan: gates via WMMA, Whh resident in VGPRs -----------
// xg: [B*T, 4H] precomputed input projections (row = b*T + t), Whh: [4H, H],
// Hout: [B*T, H]. One workgroup, 512 threads = 16 waves; wave w owns gate
// columns [32w, 32w+32). Batch B=4 folded into WMMA M rows 0..3 (4..15 zero).
// WMMA computes only h @ Whh^T (acc starts at inline 0); the xg gate bias is
// added in the elementwise phase via coalesced loads that overlap the WMMAs.
__global__ __launch_bounds__(512) void lstm_wmma(
    const float* __restrict__ xg, const float* __restrict__ Whh,
    float* __restrict__ Hout) {
  __shared__ _Float16 h_lds[16][HDIM];   // f16 hidden state, rows 4..15 = 0
  __shared__ float    g_lds[4][G4H];     // recurrent gate contributions

  int tid = threadIdx.x;
  int wave = tid >> 5, lane = tid & 31;
  int nn = lane & 15, half = lane >> 4;

  for (int i = tid; i < 16 * HDIM; i += 512) ((_Float16*)h_lds)[i] = (_Float16)0.f;

  // Preload this wave's recurrent weight slab: 2 N-tiles x 4 K-tiles = 8 v16h
  v16h Bt[2][4];
#pragma unroll
  for (int ntl = 0; ntl < 2; ++ntl)
#pragma unroll
    for (int kt = 0; kt < 4; ++kt)
      Bt[ntl][kt] = load_bT_f32(Whh, HDIM, wave * 32 + ntl * 16, kt * 32);

  int eb = tid >> 7;    // elementwise: batch 0..3
  int eh = tid & 127;   // hidden index 0..127
  float c = 0.f;
  const float* xr = xg + (size_t)(eb * TSTEPS) * G4H + eh;

  __syncthreads();

  for (int t = 0; t < TSTEPS; ++t) {
    // issue this step's xg gate-bias loads early (coalesced, overlap WMMAs)
    const float* xp = xr + (size_t)t * G4H;
    float xi = xp[0];
    float xf = xp[128];
    float xc = xp[256];
    float xo = xp[384];

    // recurrent projection: h @ Whh^T for this wave's 32 gate columns
#pragma unroll
    for (int ntl = 0; ntl < 2; ++ntl) {
      v8f acc = {};  // inline-0 C operand on first WMMA
#pragma unroll
      for (int kt = 0; kt < 4; ++kt) {
        v16h a = load_a_lds(h_lds, kt * 32);
        acc = WMMA_F16(a, Bt[ntl][kt], acc);
      }
      if (half == 0) {  // rows 0..3 are the valid batches
        int n0 = wave * 32 + ntl * 16;
#pragma unroll
        for (int r = 0; r < 4; ++r) g_lds[r][n0 + nn] = acc[r];
      }
    }
    __syncthreads();

    // gate math: i,f,g,o order
    float gi = g_lds[eb][eh]        + xi;
    float gf = g_lds[eb][128 + eh]  + xf;
    float gg = g_lds[eb][256 + eh]  + xc;
    float go = g_lds[eb][384 + eh]  + xo;
    c = sigm(gf) * c + sigm(gi) * tanh_fast(gg);
    float h = sigm(go) * tanh_fast(c);
    h_lds[eb][eh] = (_Float16)h;
    Hout[(size_t)(eb * TSTEPS + t) * HDIM + eh] = h;
    __syncthreads();
  }
}

// ---------------- fused attention: scores -> softmax -> R --------------------
// One block per (b,t). scores[s] = u . tanh(hp_t[b,t,:] + c_s[b,s,:]) + bu,
// beta = softmax_s, R[b,t,:] = sum_s beta[s] * Hsh[b,s,:].
__global__ __launch_bounds__(256) void attn_softmax_r(
    const float* __restrict__ hp_t, const float* __restrict__ c_s,
    const float* __restrict__ Hsh, const float* __restrict__ u,
    const float* __restrict__ bu, float* __restrict__ R) {
  __shared__ float ht[HDIM];
  __shared__ float uv[HDIM];
  __shared__ float sc[TSTEPS];
  __shared__ float red[256];

  int bt = blockIdx.x;
  int b = bt >> 9;  // / T
  int tid = threadIdx.x;
  if (tid < HDIM) { ht[tid] = hp_t[(size_t)bt * HDIM + tid]; uv[tid] = u[tid]; }
  __syncthreads();

  float bub = bu[0];
  float lmax = -3.4e38f;
  for (int s = tid; s < TSTEPS; s += 256) {
    const float4* cp = (const float4*)(c_s + (size_t)(b * TSTEPS + s) * HDIM);
    if (s + 256 < TSTEPS)
      __builtin_prefetch(c_s + (size_t)(b * TSTEPS + s + 256) * HDIM, 0, 0);
    float acc = 0.f;
#pragma unroll 4
    for (int h4 = 0; h4 < HDIM / 4; ++h4) {
      float4 cv = cp[h4];
      int h = h4 * 4;
      acc += uv[h + 0] * tanh_fast(ht[h + 0] + cv.x);
      acc += uv[h + 1] * tanh_fast(ht[h + 1] + cv.y);
      acc += uv[h + 2] * tanh_fast(ht[h + 2] + cv.z);
      acc += uv[h + 3] * tanh_fast(ht[h + 3] + cv.w);
    }
    float v = acc + bub;
    sc[s] = v;
    lmax = fmaxf(lmax, v);
  }
  red[tid] = lmax;
  __syncthreads();
  for (int st = 128; st > 0; st >>= 1) {
    if (tid < st) red[tid] = fmaxf(red[tid], red[tid + st]);
    __syncthreads();
  }
  float mx = red[0];
  __syncthreads();

  float lsum = 0.f;
  for (int s = tid; s < TSTEPS; s += 256) {
    float e = __expf(sc[s] - mx);
    sc[s] = e;
    lsum += e;
  }
  red[tid] = lsum;
  __syncthreads();
  for (int st = 128; st > 0; st >>= 1) {
    if (tid < st) red[tid] += red[tid + st];
    __syncthreads();
  }
  float inv = 1.f / red[0];

  if (tid < HDIM) {
    const float* hs = Hsh + (size_t)b * TSTEPS * HDIM + tid;
    float acc = 0.f;
    for (int s = 0; s < TSTEPS; ++s) acc += sc[s] * hs[(size_t)s * HDIM];
    R[(size_t)bt * HDIM + tid] = acc * inv;
  }
}

// ---------------- host orchestration ----------------------------------------
extern "C" void kernel_launch(void* const* d_in, const int* in_sizes, int n_in,
                              void* d_out, int out_size, void* d_ws, size_t ws_size,
                              hipStream_t stream) {
  (void)in_sizes; (void)n_in; (void)out_size; (void)ws_size;
  const float* x     = (const float*)d_in[0];   // [B,T,I] = [2048,128]
  const float* Wih_s = (const float*)d_in[1];   // [512,128]
  const float* Whh_s = (const float*)d_in[2];   // [512,128]
  const float* b_s   = (const float*)d_in[3];   // [512]
  const float* Wx    = (const float*)d_in[4];   // [128,128]
  const float* Wht   = (const float*)d_in[5];   // [128,128]
  const float* Whs   = (const float*)d_in[6];   // [128,128]
  const float* bs    = (const float*)d_in[7];   // [128]
  const float* u     = (const float*)d_in[8];   // [128]
  const float* bu    = (const float*)d_in[9];   // [1]
  const float* Wih_t = (const float*)d_in[10];  // [512,256]
  const float* Whh_t = (const float*)d_in[11];  // [512,128]
  const float* b_t   = (const float*)d_in[12];  // [512]
  float* out = (float*)d_out;                   // [B,T,H]

  float* ws   = (float*)d_ws;
  float* xg   = ws;                         // [2048,512] (reused for both LSTMs)
  float* Hsh  = xg + (size_t)BT_ROWS * G4H; // [2048,128]
  float* hpt  = Hsh + (size_t)BT_ROWS * HDIM;
  float* csb  = hpt + (size_t)BT_ROWS * HDIM;
  float* R    = csb + (size_t)BT_ROWS * HDIM;

  // 1) xg_s = x @ Wih_s^T + b_lstm_s                  (2048x512x128)
  gemm_wmma<true><<<256, 256, 0, stream>>>(x, Wih_s, 128, nullptr, nullptr, 0,
                                           b_s, xg, BT_ROWS, G4H, 128, 128, 0, 0, G4H);
  // 2) shared LSTM scan -> Hsh
  lstm_wmma<<<1, 512, 0, stream>>>(xg, Whh_s, Hsh);
  // 3) hp_t = Hsh @ Wht                               (2048x128x128)
  gemm_wmma<false><<<64, 256, 0, stream>>>(Hsh, Wht, 128, nullptr, nullptr, 0,
                                           nullptr, hpt, BT_ROWS, HDIM, 128, 128, 0, 0, HDIM);
  // 4) c_s = x @ Wx + Hsh @ Whs + bs                  (2048x128x(128+128))
  gemm_wmma<false><<<64, 256, 0, stream>>>(x, Wx, 128, Hsh, Whs, 128,
                                           bs, csb, BT_ROWS, HDIM, 128, 128, 128, 128, HDIM);
  // 5) fused scores/softmax/R                          (one block per (b,t))
  attn_softmax_r<<<2048, 256, 0, stream>>>(hpt, csb, Hsh, u, bu, R);
  // 6) xg_t = x @ Wih_t[:,:128]^T + R @ Wih_t[:,128:]^T + b_lstm_t
  gemm_wmma<true><<<256, 256, 0, stream>>>(x, Wih_t, 128, R, Wih_t + 128, 128,
                                           b_t, xg, BT_ROWS, G4H, 128, 256, 128, 256, G4H);
  // 7) task LSTM scan -> output
  lstm_wmma<<<1, 512, 0, stream>>>(xg, Whh_t, out);
}
